// FullAttention_27419071218249
// MI455X (gfx1250) — matrix operand order repaired and checked by reading, not compiled, not run on hardware
//
#include <hip/hip_runtime.h>
#include <hip/hip_bf16.h>

typedef __attribute__((ext_vector_type(16))) _Float16 v16h;
typedef __attribute__((ext_vector_type(8)))  _Float16 v8h;
typedef __attribute__((ext_vector_type(4)))  _Float16 v4h;
typedef __attribute__((ext_vector_type(8)))  float    v8f;

#define NB 16
#define NL 2048
#define NS 2048
#define NE 64
#define ND 64

// log2(e) / sqrt(E): fold softmax base-2 conversion into the Q scale.
#define QSCALE (0.125f * 1.4426950408889634f)

__device__ __forceinline__ v8f wmma_f16(v16h a, v16h b, v8f c) {
  return __builtin_amdgcn_wmma_f32_16x16x32_f16(
      /*neg_a=*/false, a, /*neg_b=*/false, b,
      /*c_mod=*/(short)0, c, /*reuse_a=*/false, /*reuse_b=*/false);
}

// ---- DPP16 butterfly reductions within each 16-lane half (pure VALU) ----
template <int CTRL>
__device__ __forceinline__ float dpp_mov(float x) {
  return __builtin_bit_cast(float,
      __builtin_amdgcn_mov_dpp(__builtin_bit_cast(int, x), CTRL, 0xF, 0xF, true));
}
__device__ __forceinline__ float red_max16(float x) {
  x = fmaxf(x, dpp_mov<0xB1>(x));   // quad_perm(1,0,3,2) : xor 1
  x = fmaxf(x, dpp_mov<0x4E>(x));   // quad_perm(2,3,0,1) : xor 2
  x = fmaxf(x, dpp_mov<0x141>(x));  // row_half_mirror    : xor 4
  x = fmaxf(x, dpp_mov<0x140>(x));  // row_mirror         : final cross-8 step
  return x;
}
__device__ __forceinline__ float red_add16(float x) {
  x += dpp_mov<0xB1>(x);
  x += dpp_mov<0x4E>(x);
  x += dpp_mov<0x141>(x);
  x += dpp_mov<0x140>(x);
  return x;
}

// ---------------------------------------------------------------------------
// Stage 1: flash attention.  Block = 8 waves = 128 query rows of one batch.
// S processed in chunks of 64 keys; K and V^T staged in LDS as f16, shared
// by all 8 waves.  Online softmax in registers (DPP reductions).
// ---------------------------------------------------------------------------
__global__ __launch_bounds__(256) void attn_fwd(const float* __restrict__ Q,
                                                const float* __restrict__ K,
                                                const float* __restrict__ V,
                                                float* __restrict__ O) {
  __shared__ __align__(128) _Float16 sK[64 * NE];     // [key][e]   8 KB
  __shared__ __align__(128) _Float16 sVT[ND * 64];    // [d][key]   8 KB
  __shared__ __align__(128) _Float16 sP[8 * 16 * 64]; // per-wave  16 KB

  const int tid   = threadIdx.x;
  const int b     = blockIdx.y;
  const int wave  = tid >> 5;
  const int lane  = tid & 31;
  const int half  = lane >> 4;
  const int mrow  = lane & 15;
  const int qbase = blockIdx.x * 128 + wave * 16;

  const float* Kb = K + (size_t)b * NS * NE;
  const float* Vb = V + (size_t)b * NS * ND;

  // staging coordinates (uniform trip counts, fully vectorized)
  const int kkey = tid >> 4;          // 0..15  (key within 16-key group)
  const int ke0  = (tid & 15) * 4;    // 0,4,..,60 (e offset, float4)
  const int vd   = tid & 63;          // d for V^T staging
  const int vk0  = (tid >> 6) * 16;   // 16-key group for V^T staging

  // ---- Q fragments (A layout, one per 32-wide k-chunk), pre-scaled
  const float* qrow = Q + ((size_t)b * NL + qbase + mrow) * NE;
  v16h qa[2];
#pragma unroll
  for (int kc = 0; kc < 2; ++kc) {
    const int base = kc * 32 + half * 16;
#pragma unroll
    for (int j = 0; j < 8; ++j) {
      qa[kc][j]     = (_Float16)(qrow[base + j]      * QSCALE);
      qa[kc][j + 8] = (_Float16)(qrow[base + 8 + j] * QSCALE);
    }
  }

  v8f acc[4] = {{}, {}, {}, {}};
  float rowmax[8], rowsum[8];
#pragma unroll
  for (int j = 0; j < 8; ++j) { rowmax[j] = -1e30f; rowsum[j] = 0.0f; }

  _Float16* myP = sP + wave * (16 * 64);

  for (int sc = 0; sc < NS; sc += 64) {
    __syncthreads();  // protect previous chunk's LDS reads
    // ---- K: float4 loads -> packed b64 LDS stores (row-major [key][e])
#pragma unroll
    for (int it = 0; it < 4; ++it) {
      const int key = kkey + it * 16;
      const float4 kv = *(const float4*)&Kb[(size_t)(sc + key) * NE + ke0];
      const v4h pk = {(_Float16)kv.x, (_Float16)kv.y,
                      (_Float16)kv.z, (_Float16)kv.w};
      *(v4h*)&sK[key * NE + ke0] = pk;
      __builtin_prefetch(&Kb[(size_t)(sc + 64 + key) * NE + ke0], 0, 1);
    }
    // ---- V: 16 coalesced b32 loads -> one 32B transposed LDS store [d][key]
    {
      v16h vv;
#pragma unroll
      for (int k = 0; k < 16; ++k)
        vv[k] = (_Float16)Vb[(size_t)(sc + vk0 + k) * ND + vd];
      *(v16h*)&sVT[vd * 64 + vk0] = vv;
      __builtin_prefetch(&Vb[(size_t)(sc + 64 + vk0) * ND + vd], 0, 1);
    }
    __syncthreads();

    // ---- scores: four 16x16 f32 tiles (key sub-tiles nt = 0..3)
    v8f s[4] = {{}, {}, {}, {}};
#pragma unroll
    for (int kc = 0; kc < 2; ++kc) {
#pragma unroll
      for (int nt = 0; nt < 4; ++nt) {
        const v16h kb =
            *(const v16h*)&sK[(nt * 16 + mrow) * NE + kc * 32 + half * 16];
        s[nt] = wmma_f16(qa[kc], kb, s[nt]);
      }
    }

    // ---- online softmax over the 64 new columns (base-2 exponentials)
    float p[4][8];
#pragma unroll
    for (int j = 0; j < 8; ++j) {
      float cm = fmaxf(fmaxf(s[0][j], s[1][j]), fmaxf(s[2][j], s[3][j]));
      cm = red_max16(cm);
      const float mnew = fmaxf(rowmax[j], cm);
      const float scl  = exp2f(rowmax[j] - mnew);
#pragma unroll
      for (int nt = 0; nt < 4; ++nt) p[nt][j] = exp2f(s[nt][j] - mnew);
      float rs = (p[0][j] + p[1][j]) + (p[2][j] + p[3][j]);
      rs = red_add16(rs);
      rowsum[j] = rowsum[j] * scl + rs;
      rowmax[j] = mnew;
#pragma unroll
      for (int t = 0; t < 4; ++t) acc[t][j] *= scl;
    }

    // ---- C-layout -> A-layout transpose of P via per-wave LDS scratch
#pragma unroll
    for (int j = 0; j < 8; ++j) {
      const int r = (j + 8 * half) * 64;
#pragma unroll
      for (int nt = 0; nt < 4; ++nt)
        myP[r + nt * 16 + mrow] = (_Float16)p[nt][j];
    }
    v16h pa[2];
#pragma unroll
    for (int kc = 0; kc < 2; ++kc) {
      const v8h lo = *(const v8h*)&myP[mrow * 64 + kc * 32 + half * 16];
      const v8h hi = *(const v8h*)&myP[mrow * 64 + kc * 32 + 8 + half * 16];
#pragma unroll
      for (int j = 0; j < 8; ++j) { pa[kc][j] = lo[j]; pa[kc][j + 8] = hi[j]; }
    }

    // ---- P(16x64) * Vc(64x64): 4 d-tiles x 2 k-chunks
#pragma unroll
    for (int kc = 0; kc < 2; ++kc) {
#pragma unroll
      for (int t = 0; t < 4; ++t) {
        const v16h vb =
            *(const v16h*)&sVT[(t * 16 + mrow) * 64 + kc * 32 + half * 16];
        acc[t] = wmma_f16(pa[kc], vb, acc[t]);
      }
    }
  }

  // ---- normalize and store V' (fp32) to workspace
  float* orow = O + ((size_t)b * NL + qbase) * ND;
#pragma unroll
  for (int j = 0; j < 8; ++j) {
    const float inv = 1.0f / rowsum[j];
    const int r = j + 8 * half;
#pragma unroll
    for (int t = 0; t < 4; ++t)
      orow[r * ND + t * 16 + mrow] = acc[t][j] * inv;
  }
}

// ---------------------------------------------------------------------------
// Stage 2: out[b,p,d] = sum_l mlp[p,l] * V'[b,l,d].
// k-chunks of 64; V' chunk staged transposed in LDS as shared B operand.
// ---------------------------------------------------------------------------
__global__ __launch_bounds__(256) void seq_mix(const float* __restrict__ Mx,
                                               const float* __restrict__ Vin,
                                               float* __restrict__ Out) {
  __shared__ __align__(128) _Float16 sBT[ND * 64];  // [d][l']  8 KB

  const int tid   = threadIdx.x;
  const int b     = blockIdx.y;
  const int wave  = tid >> 5;
  const int lane  = tid & 31;
  const int half  = lane >> 4;
  const int mrow  = lane & 15;
  const int pbase = blockIdx.x * 128 + wave * 16;

  const int vd  = tid & 63;
  const int vl0 = (tid >> 6) * 16;

  v8f acc[4] = {{}, {}, {}, {}};
  const float* arow = Mx + (size_t)(pbase + mrow) * NL;
  const float* Vb   = Vin + (size_t)b * NL * ND;

  for (int kk = 0; kk < NL; kk += 64) {
    __syncthreads();
    {
      v16h vv;
#pragma unroll
      for (int k = 0; k < 16; ++k)
        vv[k] = (_Float16)Vb[(size_t)(kk + vl0 + k) * ND + vd];
      *(v16h*)&sBT[vd * 64 + vl0] = vv;
      __builtin_prefetch(&Vb[(size_t)(kk + 64 + vl0) * ND + vd], 0, 1);
    }
    __syncthreads();

    v16h a[2];
#pragma unroll
    for (int kc = 0; kc < 2; ++kc) {
      const int base = kk + kc * 32 + half * 16;
#pragma unroll
      for (int j = 0; j < 8; ++j) {
        a[kc][j]     = (_Float16)arow[base + j];
        a[kc][j + 8] = (_Float16)arow[base + 8 + j];
      }
    }

#pragma unroll
    for (int kc = 0; kc < 2; ++kc) {
#pragma unroll
      for (int t = 0; t < 4; ++t) {
        const v16h bfrag =
            *(const v16h*)&sBT[(t * 16 + mrow) * 64 + kc * 32 + half * 16];
        acc[t] = wmma_f16(a[kc], bfrag, acc[t]);
      }
    }
  }

  float* orow = Out + ((size_t)b * NL + pbase) * ND;
#pragma unroll
  for (int j = 0; j < 8; ++j) {
    const int r = j + 8 * half;
#pragma unroll
    for (int t = 0; t < 4; ++t)
      orow[r * ND + t * 16 + mrow] = acc[t][j];
  }
}

extern "C" void kernel_launch(void* const* d_in, const int* in_sizes, int n_in,
                              void* d_out, int out_size, void* d_ws, size_t ws_size,
                              hipStream_t stream) {
  const float* Q  = (const float*)d_in[0];  // [B, L, E]
  const float* K  = (const float*)d_in[1];  // [B, S, E]
  const float* V  = (const float*)d_in[2];  // [B, S, D]
  const float* Mx = (const float*)d_in[3];  // [L, L]
  float* out  = (float*)d_out;              // [B, L, D]
  float* vout = (float*)d_ws;               // [B, L, D] intermediate (8 MB)

  dim3 grid(NL / 128, NB);
  attn_fwd<<<grid, 256, 0, stream>>>(Q, K, V, vout);
  seq_mix <<<grid, 256, 0, stream>>>(Mx, vout, out);
}
